// MultiHeadAttention_23691039605517
// MI455X (gfx1250) — compile-verified, matrix operand order checked by reading, not run on hardware
//
#include <hip/hip_runtime.h>
#include <hip/hip_bf16.h>
#include <math.h>

typedef __attribute__((ext_vector_type(16))) __bf16 v16bf;
typedef __attribute__((ext_vector_type(8)))  float  v8f;
typedef __attribute__((ext_vector_type(4)))  unsigned int u32x4;
typedef __attribute__((ext_vector_type(8)))  int i32x8;
typedef __attribute__((ext_vector_type(4)))  int i32x4;

#define B_     4
#define S_     2048
#define D_     768
#define H_     12
#define DEPTH_ 64
#define SCALE_ 0.125f   // 1/sqrt(64)

// ---------- fragment helpers (CDNA5 wave32 WMMA layouts) ----------
// A 16x32 bf16: lane holds row (lane&15); elements 0..7 = K[khalf..khalf+7],
// elements 8..15 = K[khalf+16..khalf+23], khalf = (lane>>4)*8.
// B 32x16 bf16: mirrored, lane holds output column n = lane&15.

__device__ __forceinline__ v16bf frag_from_f32(const float* __restrict__ p) {
  v16bf f;
#pragma unroll
  for (int i = 0; i < 8; ++i) {
    f[i]     = (__bf16)p[i];
    f[i + 8] = (__bf16)p[i + 16];
  }
  return f;
}

__device__ __forceinline__ v16bf frag_from_bf16(const __bf16* __restrict__ p) {
  v16bf f;
#pragma unroll
  for (int i = 0; i < 8; ++i) {
    f[i]     = p[i];
    f[i + 8] = p[i + 16];
  }
  return f;
}

__device__ __forceinline__ v8f wmma_bf16(v16bf a, v16bf b, v8f c) {
  // (neg_a, A, neg_b, B, c_mod, C, reuse_a, reuse_b)
  return __builtin_amdgcn_wmma_f32_16x16x32_bf16(false, a, false, b,
                                                 (short)0, c, false, false);
}

// ---------- Tensor Data Mover: 2D bf16 tile  global -> LDS ----------
// LDS aperture occupies addr[63:32]; low 32 bits of a flat shared pointer
// are the LDS byte offset.
__device__ __forceinline__ unsigned lds_off(const void* p) {
  return (unsigned)(size_t)p;
}

// D# per CDNA5 ISA ch.8: group0 = {flags, lds_addr, global_addr, type},
// group1 = {mask/data_size, tensor dims, tile dims, dim0 stride}.
// 2D tile: tile_dim0 = contiguous elements per row, tile_dim1 = rows,
// d0_stride = row pitch (elements). data_size=1 -> 2-byte elements.
__device__ __forceinline__ void tdm_load_2d_bf16(const void* lds_dst,
                                                 const void* gsrc,
                                                 unsigned tensor_d0,
                                                 unsigned tensor_d1,
                                                 unsigned tile_d0,
                                                 unsigned tile_d1,
                                                 unsigned d0_stride) {
  unsigned long long ga = (unsigned long long)(size_t)gsrc;
  u32x4 g0;
  g0[0] = 1u;                                            // count=1, user D#
  g0[1] = lds_off(lds_dst);                              // lds_addr
  g0[2] = (unsigned)(ga & 0xffffffffu);                  // global_addr[31:0]
  g0[3] = (unsigned)((ga >> 32) & 0x01ffffffu) | (2u << 30);  // [56:32] | type=2

  i32x8 g1;
  g1[0] = (int)(1u << 16);                               // data_size=1 (2B), mask=0
  g1[1] = (int)((tensor_d0 & 0xffffu) << 16);            // tensor_dim0[15:0] @ bit48
  g1[2] = (int)((tensor_d0 >> 16) | ((tensor_d1 & 0xffffu) << 16));
  g1[3] = (int)((tensor_d1 >> 16) | ((tile_d0 & 0xffffu) << 16));  // tile_dim0 @ bit112
  g1[4] = (int)(tile_d1 & 0xffffu);                      // tile_dim1, tile_dim2=0
  g1[5] = (int)d0_stride;                                // tensor_dim0_stride[31:0]
  g1[6] = 0;
  g1[7] = 0;

  i32x4 z4 = {};
#if defined(__clang_major__) && (__clang_major__ >= 23)
  i32x8 z8 = {};
  __builtin_amdgcn_tensor_load_to_lds(g0, g1, z4, z4, z8, 0);
#else
  __builtin_amdgcn_tensor_load_to_lds(g0, g1, z4, z4, 0);
#endif
}

// ---------- kernel 1: Y = X @ W^T  (fp32 in, bf16 out, head-split) ----------
// transposed==0 : y[bh][s][64]   (for Q, K)
// transposed==1 : y[bh][d][S]    (for V, so ctx B-fragments are contiguous)
__global__ void qkv_proj_kernel(const float* __restrict__ x,
                                const float* __restrict__ w,
                                __bf16* __restrict__ y,
                                int transposed) {
  const int wave  = threadIdx.x >> 5;
  const int lane  = threadIdx.x & 31;
  const int half  = lane >> 4;
  const int khalf = half * 8;
  const int nrow  = lane & 15;

  const int job = blockIdx.x * 4 + wave;        // 0 .. 6143
  const int mt  = job / (D_ / DEPTH_);          // 16-row tile over B*S
  const int nt  = job % (D_ / DEPTH_);          // head index
  const int m0  = mt * 16;

  v8f zero = {};
  v8f acc[4];
#pragma unroll
  for (int t = 0; t < 4; ++t) acc[t] = zero;

  for (int kb = 0; kb < D_; kb += 32) {
    v16bf a = frag_from_f32(x + (size_t)(m0 + nrow) * D_ + kb + khalf);
#pragma unroll
    for (int t = 0; t < 4; ++t) {
      v16bf b = frag_from_f32(w + (size_t)(nt * DEPTH_ + t * 16 + nrow) * D_ + kb + khalf);
      acc[t] = wmma_bf16(a, b, acc[t]);
    }
  }

  const int    bb = m0 / S_;
  const int    s0 = m0 % S_;
  const size_t bh = (size_t)bb * H_ + nt;

  if (!transposed) {
#pragma unroll
    for (int t = 0; t < 4; ++t)
#pragma unroll
      for (int r = 0; r < 8; ++r) {
        int M = r + 8 * half;
        y[(bh * S_ + s0 + M) * DEPTH_ + t * 16 + nrow] = (__bf16)acc[t][r];
      }
  } else {
#pragma unroll
    for (int t = 0; t < 4; ++t)
#pragma unroll
      for (int r = 0; r < 8; ++r) {
        int M = r + 8 * half;
        int d = t * 16 + nrow;
        y[(bh * DEPTH_ + d) * S_ + s0 + M] = (__bf16)acc[t][r];
      }
  }
}

// ---------- kernel 2: flash attention with TDM-staged K/V tiles ----------
__global__ void flash_attn_kernel(const __bf16* __restrict__ qh,
                                  const __bf16* __restrict__ kh,
                                  const __bf16* __restrict__ vt,
                                  __bf16* __restrict__ ctx,
                                  float* __restrict__ m0_ws,
                                  float* __restrict__ l0_ws) {
  __shared__ __bf16 kbuf[2][32][64];   // 32 keys x 64 depth, double-buffered (8KB)
  __shared__ __bf16 vbuf[2][64][32];   // 64 depth x 32 keys, double-buffered (8KB)
  __shared__ __bf16 pbuf[4][16][32];   // per-wave P transpose buffer (4KB)

  const int wave  = threadIdx.x >> 5;
  const int lane  = threadIdx.x & 31;
  const int half  = lane >> 4;
  const int khalf = half * 8;
  const int nrow  = lane & 15;

  const int bh  = blockIdx.x >> 5;     // 0..47
  const int q64 = blockIdx.x & 31;
  const int q0  = q64 * 64 + wave * 16;

  const __bf16* qbase = qh + ((size_t)bh * S_ + q0) * DEPTH_;
  const __bf16* kbase = kh + (size_t)bh * S_ * DEPTH_;
  const __bf16* vbase = vt + (size_t)bh * DEPTH_ * S_;

  // Q fragments held in registers for the whole pass (depth 64 = 2 K-chunks)
  v16bf aq0 = frag_from_bf16(qbase + (size_t)nrow * DEPTH_ + 0  + khalf);
  v16bf aq1 = frag_from_bf16(qbase + (size_t)nrow * DEPTH_ + 32 + khalf);

  v8f zero = {};
  v8f acc[4];
  float m[8], l[8];
#pragma unroll
  for (int t = 0; t < 4; ++t) acc[t] = zero;
#pragma unroll
  for (int r = 0; r < 8; ++r) { m[r] = -3.0e38f; l[r] = 0.0f; }

  const int NB = S_ / 32;  // 64 key blocks

  // prologue: TDM-load block 0 into buffer 0 (one DMA pair per workgroup)
  if (wave == 0) {
    tdm_load_2d_bf16(&kbuf[0][0][0], kbase, DEPTH_, S_, DEPTH_, 32, DEPTH_);
    tdm_load_2d_bf16(&vbuf[0][0][0], vbase, S_, DEPTH_, 32, DEPTH_, S_);
  }

  for (int ib = 0; ib < NB; ++ib) {
    const int k0  = ib * 32;
    const int cur = ib & 1;
    const int nxt = cur ^ 1;

    if (wave == 0) {
      if (ib + 1 < NB) {
        // issue next block's DMAs, then wait for current block (TDM is
        // in-order per wave: <=2 outstanding means block ib completed)
        const int kn = k0 + 32;
        tdm_load_2d_bf16(&kbuf[nxt][0][0], kbase + (size_t)kn * DEPTH_,
                         DEPTH_, S_, DEPTH_, 32, DEPTH_);
        tdm_load_2d_bf16(&vbuf[nxt][0][0], vbase + kn,
                         S_, DEPTH_, 32, DEPTH_, S_);
        __builtin_amdgcn_s_wait_tensorcnt(2);
      } else {
        __builtin_amdgcn_s_wait_tensorcnt(0);
      }
    }
    __syncthreads();   // kbuf[cur]/vbuf[cur] visible to all 4 waves

    // ---- logits: 16 queries x 32 keys, two 16x16 C tiles, from LDS ----
    v8f c0 = zero, c1 = zero;
    c0 = wmma_bf16(aq0, frag_from_bf16(&kbuf[cur][nrow][khalf]),           c0);
    c0 = wmma_bf16(aq1, frag_from_bf16(&kbuf[cur][nrow][32 + khalf]),      c0);
    c1 = wmma_bf16(aq0, frag_from_bf16(&kbuf[cur][16 + nrow][khalf]),      c1);
    c1 = wmma_bf16(aq1, frag_from_bf16(&kbuf[cur][16 + nrow][32 + khalf]), c1);

    // ---- online softmax update (row stats via 16-lane shfl reductions) ----
#pragma unroll
    for (int r = 0; r < 8; ++r) {
      float v0 = c0[r] * SCALE_, v1 = c1[r] * SCALE_;
      float mx = fmaxf(v0, v1);
#pragma unroll
      for (int off = 8; off >= 1; off >>= 1) mx = fmaxf(mx, __shfl_xor(mx, off, 32));
      float mn   = fmaxf(m[r], mx);
      float corr = __expf(m[r] - mn);
      float p0   = __expf(v0 - mn);
      float p1   = __expf(v1 - mn);
      float sum  = p0 + p1;
#pragma unroll
      for (int off = 8; off >= 1; off >>= 1) sum += __shfl_xor(sum, off, 32);
      l[r] = l[r] * corr + sum;
      m[r] = mn;
#pragma unroll
      for (int t = 0; t < 4; ++t) acc[t][r] *= corr;
      c0[r] = p0; c1[r] = p1;
    }

    // ---- transpose P (C layout -> A layout) through per-wave LDS ----
#pragma unroll
    for (int r = 0; r < 8; ++r) {
      int M = r + 8 * half;
      pbuf[wave][M][nrow]      = (__bf16)c0[r];
      pbuf[wave][M][nrow + 16] = (__bf16)c1[r];
    }
    asm volatile("s_wait_dscnt 0" ::: "memory");  // DS in-order + explicit wait
    v16bf ap = frag_from_bf16(&pbuf[wave][nrow][khalf]);

    // ---- ctx += P(16x32) x V(32x64), V fragments from LDS ----
#pragma unroll
    for (int t = 0; t < 4; ++t) {
      v16bf bv = frag_from_bf16(&vbuf[cur][t * 16 + nrow][khalf]);
      acc[t] = wmma_bf16(ap, bv, acc[t]);
    }

    __syncthreads();   // all waves done with buffer `cur`; TDM may overwrite
  }

  // ---- normalize and store ctx (bf16, merged-head layout [B*S, D]) ----
  const int bb = bh / H_, hh = bh % H_;
#pragma unroll
  for (int r = 0; r < 8; ++r) {
    int   M   = r + 8 * half;
    float inv = 1.0f / l[r];
#pragma unroll
    for (int t = 0; t < 4; ++t) {
      ctx[((size_t)bb * S_ + q0 + M) * D_ + hh * DEPTH_ + t * 16 + nrow] =
          (__bf16)(acc[t][r] * inv);
    }
    if (hh == 0 && nrow == 0) {
      m0_ws[bb * S_ + q0 + M] = m[r];
      l0_ws[bb * S_ + q0 + M] = l[r];
    }
  }
}

// ---------- kernel 3: head-0 attention probabilities (second output) ----------
__global__ void attn_probs_kernel(const __bf16* __restrict__ qh,
                                  const __bf16* __restrict__ kh,
                                  const float* __restrict__ m0_ws,
                                  const float* __restrict__ l0_ws,
                                  float* __restrict__ attn) {
  const int wave  = threadIdx.x >> 5;
  const int lane  = threadIdx.x & 31;
  const int half  = lane >> 4;
  const int khalf = half * 8;
  const int nrow  = lane & 15;

  const int job = blockIdx.x * 4 + wave;  // 0..511
  const int bb  = job >> 7;
  const int q0  = (job & 127) * 16;

  const __bf16* qbase = qh + ((size_t)bb * H_ * S_ + q0) * DEPTH_;  // head 0
  const __bf16* kbase = kh + (size_t)bb * H_ * S_ * DEPTH_;

  v16bf aq0 = frag_from_bf16(qbase + (size_t)nrow * DEPTH_ + 0  + khalf);
  v16bf aq1 = frag_from_bf16(qbase + (size_t)nrow * DEPTH_ + 32 + khalf);

  float mv[8], il[8];
#pragma unroll
  for (int r = 0; r < 8; ++r) {
    int M = r + 8 * half;
    mv[r] = m0_ws[bb * S_ + q0 + M];
    il[r] = 1.0f / l0_ws[bb * S_ + q0 + M];
  }

  v8f zero = {};
  for (int k0 = 0; k0 < S_; k0 += 16) {
    v8f c = zero;
    const __bf16* kp = kbase + (size_t)(k0 + nrow) * DEPTH_;
    c = wmma_bf16(aq0, frag_from_bf16(kp + khalf),      c);
    c = wmma_bf16(aq1, frag_from_bf16(kp + 32 + khalf), c);
#pragma unroll
    for (int r = 0; r < 8; ++r) {
      int M = r + 8 * half;
      attn[((size_t)bb * S_ + q0 + M) * S_ + k0 + nrow] =
          __expf(c[r] * SCALE_ - mv[r]) * il[r];
    }
  }
}

// ---------- kernel 4: out = ctx @ wd^T + bd ----------
__global__ void out_proj_kernel(const __bf16* __restrict__ ctx,
                                const float* __restrict__ wd,
                                const float* __restrict__ bd,
                                float* __restrict__ out) {
  const int wave  = threadIdx.x >> 5;
  const int lane  = threadIdx.x & 31;
  const int half  = lane >> 4;
  const int khalf = half * 8;
  const int nrow  = lane & 15;

  const int job = blockIdx.x * 4 + wave;   // 0..6143
  const int m0  = (job / (D_ / DEPTH_)) * 16;
  const int n0  = (job % (D_ / DEPTH_)) * DEPTH_;

  v8f zero = {};
  v8f acc[4];
#pragma unroll
  for (int t = 0; t < 4; ++t) acc[t] = zero;

  for (int kb = 0; kb < D_; kb += 32) {
    v16bf a = frag_from_bf16(ctx + (size_t)(m0 + nrow) * D_ + kb + khalf);
#pragma unroll
    for (int t = 0; t < 4; ++t) {
      v16bf b = frag_from_f32(wd + (size_t)(n0 + t * 16 + nrow) * D_ + kb + khalf);
      acc[t] = wmma_bf16(a, b, acc[t]);
    }
  }

#pragma unroll
  for (int t = 0; t < 4; ++t) {
    float bias = bd[n0 + t * 16 + nrow];
#pragma unroll
    for (int r = 0; r < 8; ++r) {
      int M = r + 8 * half;
      out[(size_t)(m0 + M) * D_ + n0 + t * 16 + nrow] = acc[t][r] + bias;
    }
  }
}

// ---------- host launch ----------
extern "C" void kernel_launch(void* const* d_in, const int* in_sizes, int n_in,
                              void* d_out, int out_size, void* d_ws, size_t ws_size,
                              hipStream_t stream) {
  (void)in_sizes; (void)n_in; (void)out_size; (void)ws_size;

  const float* q  = (const float*)d_in[0];
  const float* k  = (const float*)d_in[1];
  const float* v  = (const float*)d_in[2];
  const float* wq = (const float*)d_in[3];
  const float* wk = (const float*)d_in[4];
  const float* wv = (const float*)d_in[5];
  const float* wd = (const float*)d_in[6];
  const float* bd = (const float*)d_in[7];

  float* out  = (float*)d_out;
  float* attn = out + (size_t)B_ * S_ * D_;   // second output region

  char*  ws  = (char*)d_ws;
  size_t per = (size_t)B_ * H_ * S_ * DEPTH_ * sizeof(__bf16);  // 12.58 MB
  __bf16* qh  = (__bf16*)(ws);
  __bf16* kh  = (__bf16*)(ws + per);
  __bf16* vt  = (__bf16*)(ws + 2 * per);
  __bf16* ctx = (__bf16*)(ws + 3 * per);
  float*  m0w = (float*)(ws + 4 * per);
  float*  l0w = m0w + B_ * S_;

  dim3 blk(128);  // 4 wave32s
  const int gemm_blocks = ((B_ * S_ / 16) * (D_ / DEPTH_)) / 4;  // 1536

  qkv_proj_kernel<<<gemm_blocks, blk, 0, stream>>>(q, wq, qh, 0);
  qkv_proj_kernel<<<gemm_blocks, blk, 0, stream>>>(k, wk, kh, 0);
  qkv_proj_kernel<<<gemm_blocks, blk, 0, stream>>>(v, wv, vt, 1);
  flash_attn_kernel<<<B_ * H_ * (S_ / 64), blk, 0, stream>>>(qh, kh, vt, ctx, m0w, l0w);
  attn_probs_kernel<<<(B_ * (S_ / 16)) / 4, blk, 0, stream>>>(qh, kh, m0w, l0w, attn);
  out_proj_kernel<<<gemm_blocks, blk, 0, stream>>>(ctx, wd, bd, out);
}
